// MultiHeadRotaryAttention_34823594836483
// MI455X (gfx1250) — matrix operand order repaired and checked by reading, not compiled, not run on hardware
//
#include <hip/hip_runtime.h>

#define D_HID 1024
#define NHEADS 16
#define HD 64
#define SEQ 2048
#define BATCH 2
#define MTOT (BATCH * SEQ)  // 4096

typedef __attribute__((ext_vector_type(16))) __bf16 v16bf;
typedef __attribute__((ext_vector_type(8)))  float  v8f;
typedef __attribute__((ext_vector_type(2)))  float  v2f;
typedef __attribute__((ext_vector_type(4)))  unsigned int u32x4;
typedef __attribute__((ext_vector_type(8)))  int i32x8;
typedef __attribute__((ext_vector_type(4)))  int i32x4;

__device__ __forceinline__ unsigned short f2bf(float x) {
  unsigned int u = __float_as_uint(x);
  u += 0x7fffu + ((u >> 16) & 1u);      // round-to-nearest-even
  return (unsigned short)(u >> 16);
}
__device__ __forceinline__ float bf2f(unsigned short h) {
  return __uint_as_float(((unsigned int)h) << 16);
}

union FragBF { v16bf v; uint4 q[2]; };

// ---------------------------------------------------------------------------
// TDM: async 2-D bf16 tile load Global -> LDS with hardware row padding.
// D# per CDNA5 ISA ch.8: group0 {count,lds,global,type}, group1 {dims/strides}.
// pad_interval=3 (every 16 DWORDs = one 32-elem bf16 row), pad_amount=3
// (4 DWORDs) -> LDS pitch 40 u16 = 80 B, matching the fragment layout.
// Toolchain is the 6-arg builtin: (g0, g1, g2, g3, g4, cpol).
// ---------------------------------------------------------------------------
__device__ __forceinline__ void tdm_load_tile_bf16(
    unsigned lds_off, unsigned long long gaddr_bytes,
    unsigned tensor_d0, unsigned tensor_d1,
    unsigned tile_d0, unsigned tile_d1, unsigned stride0) {
  u32x4 g0;
  g0[0] = 1u;                                          // count=1 (valid, user)
  g0[1] = lds_off;                                     // lds_addr [63:32]
  g0[2] = (unsigned)(gaddr_bytes & 0xffffffffu);       // global_addr [95:64]
  g0[3] = (unsigned)((gaddr_bytes >> 32) & 0x1ffffffu) // global_addr [120:96]
          | (2u << 30);                                // type=2 ("image")
  i32x8 g1;
  g1[0] = (int)((1u << 16)      // data_size = 1 -> 2 bytes
              | (1u << 20)      // pad_enable
              | (3u << 22)      // pad_interval: 16 DWORDs
              | (3u << 25));    // pad_amount:   4 DWORDs
  g1[1] = (int)((tensor_d0 & 0xffffu) << 16);                        // td0 lo
  g1[2] = (int)(((tensor_d0 >> 16) & 0xffffu) |
                ((tensor_d1 & 0xffffu) << 16));                      // td0hi|td1lo
  g1[3] = (int)(((tensor_d1 >> 16) & 0xffffu) |
                ((tile_d0 & 0xffffu) << 16));                        // td1hi|tile0
  g1[4] = (int)(tile_d1 & 0xffffu);                                  // tile1, tile2=0
  g1[5] = (int)stride0;                                              // dim0 stride lo
  g1[6] = 0;                                                         // stride hi | s1
  g1[7] = 0;
  const i32x4 z4 = {0, 0, 0, 0};
  const i32x8 z8 = {0, 0, 0, 0, 0, 0, 0, 0};
  __builtin_amdgcn_tensor_load_to_lds(g0, g1, z4, z4, z8, 0);
}

// ---------------------------------------------------------------------------
// Prepass: split f32 weight [K][N] into transposed bf16 hi/lo planes [N][K].
// Amortized once; planes (16 MB total) stay resident in the 192 MB L2.
// ---------------------------------------------------------------------------
__global__ __launch_bounds__(256)
void split_transpose(const float* __restrict__ W, unsigned short* __restrict__ hi,
                     unsigned short* __restrict__ lo, int K, int N) {
  const int idx = blockIdx.x * blockDim.x + threadIdx.x;
  if (idx >= K * N) return;
  const int k = idx / N;
  const int n = idx - k * N;
  const float x = W[idx];
  const unsigned short h = f2bf(x);
  hi[(size_t)n * K + k] = h;
  lo[(size_t)n * K + k] = f2bf(x - bf2f(h));
}

// ---------------------------------------------------------------------------
// C[M,N] = A[M,K] @ W[K,N] + bias  via bf16x3 split WMMA.
// A (f32) staged through VGPRs (the hi/lo split needs VALU anyway);
// W tiles pulled by the Tensor Data Mover from pre-split planes.
// Block tile 128x64, K-step 32, 8 waves, each wave 32x32.
// ---------------------------------------------------------------------------
#define BM 128
#define BN 64
#define BK 32
#define APAD 40  // u16 row pitch: 80 B -> 16B-aligned b128 fragment loads

__global__ __launch_bounds__(256)
void gemm_bias_bf16x3(const float* __restrict__ A,
                      const unsigned short* __restrict__ WThi,
                      const unsigned short* __restrict__ WTlo,
                      const float* __restrict__ bias, float* __restrict__ C,
                      int M, int N, int K) {
  __shared__ __align__(16) unsigned short Ahi[BM][APAD];
  __shared__ __align__(16) unsigned short Alo[BM][APAD];
  __shared__ __align__(16) unsigned short Bhi[BN][APAD];  // [n][k], TDM-filled
  __shared__ __align__(16) unsigned short Blo[BN][APAD];

  const int t    = threadIdx.x;
  const int m0   = blockIdx.y * BM;
  const int n0   = blockIdx.x * BN;
  const int lane = t & 31;
  const int wid  = t >> 5;
  const int wm   = (wid & 3) * 32;
  const int wn   = (wid >> 2) * 32;
  const int h15  = lane & 15;
  const int half = lane >> 4;
  const int abase = half * 8;   // A frag: K = {abase..+7} U {16+abase..+7}
  const int bbase = half * 8;   // B frag: K = {bbase..+7} U {16+bbase..+7}

  const int a_col = (t & 7) * 4;
  const int a_row = t >> 3;

  const unsigned ldsBhi = (unsigned)(size_t)&Bhi[0][0];
  const unsigned ldsBlo = (unsigned)(size_t)&Blo[0][0];
  const unsigned long long gHi = (unsigned long long)(size_t)WThi;
  const unsigned long long gLo = (unsigned long long)(size_t)WTlo;

  v8f acc[2][2] = {};

  for (int kc = 0; kc < K; kc += BK) {
    // ---- async TDM: B tiles (64 rows x 32 bf16) from transposed planes ----
    if (wid == 0) {
      const unsigned long long off = ((size_t)n0 * K + kc) * 2ull;
      tdm_load_tile_bf16(ldsBhi, gHi + off, K, N, BK, BN, K);
      tdm_load_tile_bf16(ldsBlo, gLo + off, K, N, BK, BN, K);
    }
    // ---- stage A tile (128x32 f32 -> bf16 hi/lo) through VGPRs ----
#pragma unroll
    for (int p = 0; p < 4; ++p) {
      const int r = a_row + 32 * p;
      const float* src = A + (size_t)(m0 + r) * K + kc + a_col;
      if (kc + BK < K) __builtin_prefetch(src + BK, 0, 1);
      const float4 x = *(const float4*)src;
      const float xs[4] = {x.x, x.y, x.z, x.w};
      unsigned short hh[4], ll[4];
#pragma unroll
      for (int i = 0; i < 4; ++i) {
        hh[i] = f2bf(xs[i]);
        ll[i] = f2bf(xs[i] - bf2f(hh[i]));
      }
      *(uint2*)&Ahi[r][a_col] =
          make_uint2((unsigned)hh[0] | ((unsigned)hh[1] << 16),
                     (unsigned)hh[2] | ((unsigned)hh[3] << 16));
      *(uint2*)&Alo[r][a_col] =
          make_uint2((unsigned)ll[0] | ((unsigned)ll[1] << 16),
                     (unsigned)ll[2] | ((unsigned)ll[3] << 16));
    }
    if (wid == 0) __builtin_amdgcn_s_wait_tensorcnt(0);
    __syncthreads();

    FragBF ah[2], al[2], bh[2], bl[2];
#pragma unroll
    for (int i = 0; i < 2; ++i) {
      const int m = wm + i * 16 + h15;
      ah[i].q[0] = *(const uint4*)&Ahi[m][abase];
      ah[i].q[1] = *(const uint4*)&Ahi[m][16 + abase];
      al[i].q[0] = *(const uint4*)&Alo[m][abase];
      al[i].q[1] = *(const uint4*)&Alo[m][16 + abase];
      const int n = wn + i * 16 + h15;
      bh[i].q[0] = *(const uint4*)&Bhi[n][bbase];
      bh[i].q[1] = *(const uint4*)&Bhi[n][bbase + 16];
      bl[i].q[0] = *(const uint4*)&Blo[n][bbase];
      bl[i].q[1] = *(const uint4*)&Blo[n][bbase + 16];
    }
#pragma unroll
    for (int i = 0; i < 2; ++i)
#pragma unroll
      for (int j = 0; j < 2; ++j) {
        acc[i][j] = __builtin_amdgcn_wmma_f32_16x16x32_bf16(
            false, ah[i].v, false, bh[j].v, (short)0, acc[i][j], false, false);
        acc[i][j] = __builtin_amdgcn_wmma_f32_16x16x32_bf16(
            false, ah[i].v, false, bl[j].v, (short)0, acc[i][j], false, false);
        acc[i][j] = __builtin_amdgcn_wmma_f32_16x16x32_bf16(
            false, al[i].v, false, bh[j].v, (short)0, acc[i][j], false, false);
      }
    __syncthreads();
  }

  // ---- epilogue: bias add; C layout VGPR r -> M = r + 8*half, N = lane&15
#pragma unroll
  for (int j = 0; j < 2; ++j) {
    const int col = n0 + wn + j * 16 + h15;
    const float bv = bias[col];
#pragma unroll
    for (int i = 0; i < 2; ++i) {
#pragma unroll
      for (int r = 0; r < 8; ++r) {
        const int row = m0 + wm + i * 16 + r + (half << 3);
        C[(size_t)row * N + col] = acc[i][j][r] + bv;
      }
    }
  }
}

// ---------------------------------------------------------------------------
// In-place RoPE over full hidden dim. Each thread owns the (j, j+512) pair.
// ---------------------------------------------------------------------------
__global__ __launch_bounds__(256)
void rope_inplace(float* __restrict__ x, int total) {
  const int idx = blockIdx.x * blockDim.x + threadIdx.x;
  if (idx >= total) return;
  const int j   = idx & 511;
  const int tok = idx >> 9;
  const int s   = tok & (SEQ - 1);
  // inv_freq[j] = 10000^(-2j/1024) = 2^(-j/512 * log2(10000))
  const float inv_freq = exp2f(-(float)j * (13.287712379549449f / 512.0f));
  const float ang = (float)s * inv_freq;
  float sn, cs;
  sincosf(ang, &sn, &cs);
  float* p = x + (size_t)tok * D_HID;
  const float x1 = p[j];
  const float x2 = p[j + 512];
  p[j]       = x1 * cs - x2 * sn;
  p[j + 512] = x2 * cs + x1 * sn;
}

// ---------------------------------------------------------------------------
// Per-token 16x16 attention (faithful to source): one wave32 per token.
// Exact fp32 WMMA (16x16x4). ctx written in transpose(0,2,1,3)+reshape order.
// ---------------------------------------------------------------------------
__global__ __launch_bounds__(256)
void attn_rotary(const float* __restrict__ q, const float* __restrict__ k,
                 const float* __restrict__ v, float* __restrict__ attn_out,
                 float* __restrict__ ctx) {
  __shared__ float attn_s[8][16][18];
  const int lane  = threadIdx.x & 31;
  const int w     = threadIdx.x >> 5;
  const int token = blockIdx.x * 8 + w;
  const int h15   = lane & 15;
  const int half  = lane >> 4;

  const float* qp = q + (size_t)token * D_HID;
  const float* kp = k + (size_t)token * D_HID;
  const float* vp = v + (size_t)token * D_HID;

  v8f c = {};
#pragma unroll
  for (int kk0 = 0; kk0 < HD; kk0 += 4) {
    const int ko = kk0 + 2 * half;
    const v2f a = *(const v2f*)(qp + h15 * HD + ko);  // q[head][dim]
    const v2f b = *(const v2f*)(kp + h15 * HD + ko);  // k^T[dim][head]
    c = __builtin_amdgcn_wmma_f32_16x16x4_f32(false, a, false, b, (short)0, c,
                                              false, false);
  }

#pragma unroll
  for (int r = 0; r < 8; ++r) {
    const float x = c[r] * 0.125f;  // 1/sqrt(64)
    float mx = x;
    mx = fmaxf(mx, __shfl_xor(mx, 1, 32));
    mx = fmaxf(mx, __shfl_xor(mx, 2, 32));
    mx = fmaxf(mx, __shfl_xor(mx, 4, 32));
    mx = fmaxf(mx, __shfl_xor(mx, 8, 32));
    float e = __expf(x - mx);
    float ssum = e;
    ssum += __shfl_xor(ssum, 1, 32);
    ssum += __shfl_xor(ssum, 2, 32);
    ssum += __shfl_xor(ssum, 4, 32);
    ssum += __shfl_xor(ssum, 8, 32);
    const float aout = e / ssum;
    const int mrow = r + (half << 3);
    attn_out[(size_t)token * 256 + mrow * 16 + h15] = aout;  // [B,S,H,H]
    attn_s[w][mrow][h15] = aout;
  }
  __syncthreads();

  const int b = token / SEQ;
  const int s = token % SEQ;
  float* cb = ctx + (size_t)b * SEQ * D_HID;
#pragma unroll
  for (int nc = 0; nc < 4; ++nc) {
    v8f d = {};
#pragma unroll
    for (int kk0 = 0; kk0 < 16; kk0 += 4) {
      const int ko = kk0 + 2 * half;
      const v2f a = *(const v2f*)&attn_s[w][h15][ko];
      v2f bb;
      bb.x = vp[(ko + 0) * HD + nc * 16 + h15];
      bb.y = vp[(ko + 1) * HD + nc * 16 + h15];
      d = __builtin_amdgcn_wmma_f32_16x16x4_f32(false, a, false, bb, (short)0,
                                                d, false, false);
    }
#pragma unroll
    for (int r = 0; r < 8; ++r) {
      const int mrow = r + (half << 3);
      const int j = nc * 16 + h15;
      cb[(size_t)mrow * SEQ * HD + (size_t)s * HD + j] = d[r];  // h*S*hd+s*hd+j
    }
  }
}

// ---------------------------------------------------------------------------
extern "C" void kernel_launch(void* const* d_in, const int* in_sizes, int n_in,
                              void* d_out, int out_size, void* d_ws,
                              size_t ws_size, hipStream_t stream) {
  const float* inputs  = (const float*)d_in[0];
  const float* context = (const float*)d_in[1];
  const float* Wq = (const float*)d_in[2];
  const float* bq = (const float*)d_in[3];
  const float* Wk = (const float*)d_in[4];
  const float* bk = (const float*)d_in[5];
  const float* Wv = (const float*)d_in[6];
  const float* bv = (const float*)d_in[7];
  const float* Wo = (const float*)d_in[8];
  const float* bo = (const float*)d_in[9];

  float* out = (float*)d_out;             // [B,S,D] then [B,S,H,H]
  float* ws  = (float*)d_ws;
  const size_t Q = (size_t)MTOT * D_HID;  // 4194304 floats
  float* qb = ws;
  float* kb = ws + Q;
  float* vb = ws + 2 * Q;
  float* cx = ws + 3 * Q;

  const size_t WSZ = (size_t)D_HID * D_HID;  // elements per weight plane
  unsigned short* wp = (unsigned short*)(ws + 4 * Q);
  unsigned short* WqH = wp + 0 * WSZ; unsigned short* WqL = wp + 1 * WSZ;
  unsigned short* WkH = wp + 2 * WSZ; unsigned short* WkL = wp + 3 * WSZ;
  unsigned short* WvH = wp + 4 * WSZ; unsigned short* WvL = wp + 5 * WSZ;
  unsigned short* WoH = wp + 6 * WSZ; unsigned short* WoL = wp + 7 * WSZ;

  const int sgrid = (int)(WSZ / 256);
  split_transpose<<<sgrid, 256, 0, stream>>>(Wq, WqH, WqL, D_HID, D_HID);
  split_transpose<<<sgrid, 256, 0, stream>>>(Wk, WkH, WkL, D_HID, D_HID);
  split_transpose<<<sgrid, 256, 0, stream>>>(Wv, WvH, WvL, D_HID, D_HID);
  split_transpose<<<sgrid, 256, 0, stream>>>(Wo, WoH, WoL, D_HID, D_HID);

  const dim3 gg(D_HID / BN, MTOT / BM);  // (16, 32)
  gemm_bias_bf16x3<<<gg, 256, 0, stream>>>(inputs,  WqH, WqL, bq, qb, MTOT, D_HID, D_HID);
  gemm_bias_bf16x3<<<gg, 256, 0, stream>>>(context, WkH, WkL, bk, kb, MTOT, D_HID, D_HID);
  gemm_bias_bf16x3<<<gg, 256, 0, stream>>>(context, WvH, WvL, bv, vb, MTOT, D_HID, D_HID);

  const int rope_total = MTOT * (D_HID / 2);
  rope_inplace<<<rope_total / 256, 256, 0, stream>>>(qb, rope_total);
  rope_inplace<<<rope_total / 256, 256, 0, stream>>>(kb, rope_total);

  attn_rotary<<<MTOT / 8, 256, 0, stream>>>(qb, kb, vb, out + Q, cx);

  gemm_bias_bf16x3<<<gg, 256, 0, stream>>>(cx, WoH, WoL, bo, out, MTOT, D_HID, D_HID);
}
